// SpatialTransformer_80195629351200
// MI455X (gfx1250) — compile-verified, matrix operand order checked by reading, not run on hardware
//
#include <hip/hip_runtime.h>

typedef _Float16 h8  __attribute__((ext_vector_type(8)));
typedef _Float16 h16 __attribute__((ext_vector_type(16)));
typedef float    f8  __attribute__((ext_vector_type(8)));

#define CMID 128
#define KTOT 1152   // 9 * 128, K ordered as (ky*3+kx)*128 + ci
#define PADW 136    // 128 ci + 8-half pad -> 272B pixel stride, conflict-free LDS

// ---------------------------------------------------------------------------
// conv0: 1x1 conv (16 -> 128 ch), NCHW f32 input -> NHWC f32 output (+bias)
// ---------------------------------------------------------------------------
__global__ __launch_bounds__(256) void k_conv0(
    const float* __restrict__ x, const float* __restrict__ w0,
    const float* __restrict__ b0, float* __restrict__ out, int total)
{
  int idx = blockIdx.x * 256 + threadIdx.x;
  if (idx >= total) return;
  int co = idx & 127;
  int p  = idx >> 7;                    // n*224*224 + h*224 + w
  int n  = p / (224 * 224);
  int hw = p - n * (224 * 224);
  const float* xp = x + (size_t)n * 16 * 224 * 224 + hw;
  float s = b0[co];
#pragma unroll
  for (int ci = 0; ci < 16; ++ci)
    s += xp[(size_t)ci * 224 * 224] * w0[co * 16 + ci];
  out[idx] = s;
}

// ---------------------------------------------------------------------------
// weight prep: OIHW f32 -> f16 [stage][co][(ky*3+kx)*128 + ci]
// ---------------------------------------------------------------------------
__global__ __launch_bounds__(256) void k_prep_w(
    const float* __restrict__ w, _Float16* __restrict__ wt, int total)
{
  int idx = blockIdx.x * 256 + threadIdx.x;
  if (idx >= total) return;
  int k   = idx % KTOT;
  int t   = idx / KTOT;
  int co  = t % 128;
  int st  = t / 128;
  int kpos = k >> 7;                 // 0..8
  int ci   = k & 127;
  int ky   = kpos / 3, kx = kpos - ky * 3;
  size_t src = ((((size_t)(st * 128 + co)) * 128 + ci) * 3 + ky) * 3 + kx;
  wt[idx] = (_Float16)w[src];
}

// ---------------------------------------------------------------------------
// 3x3 valid conv, implicit GEMM via WMMA f16 -> f32.
// act: NHWC f16 [N][Hin][Win][128].  wt: [128][1152] f16.
// out: NHWC f32 [N][Ho][Wo][128] (+bias).
// Block = 256 thr = 8 waves; the block stages the 3x18x128 input patch for
// one 16-pixel output tile in LDS (padded to kill bank conflicts); wave w
// handles co in [16w,16w+16). Per WMMA: A = 2 global b128 (weights, L0/L2
// resident), B = 2 ds b128 from LDS. A/B lane fragments follow the ISA
// 16-bit layout (two contiguous 8-half groups at K offsets half*8 and
// 16+half*8 within the 32-wide chunk).
// ---------------------------------------------------------------------------
__global__ __launch_bounds__(256) void k_conv3x3_wmma(
    const _Float16* __restrict__ act, const _Float16* __restrict__ wt,
    const float* __restrict__ bias, float* __restrict__ out,
    int Hin, int Win, int Ho, int Wo)
{
  __shared__ __align__(16) _Float16 patch[3 * 18 * PADW];

  const int tid    = threadIdx.x;
  const int lane   = tid & 31;
  const int hf     = lane >> 4;              // 0/1
  const int lm     = lane & 15;              // A: co row, B: pixel col
  const int cobase = (tid >> 5) * 16;

  const int tpr = (Wo + 15) >> 4;
  int tile = blockIdx.x;
  const int n = tile / (Ho * tpr);
  int r = tile - n * Ho * tpr;
  const int oy = r / tpr;
  const int oxb = (r - oy * tpr) << 4;

  // ---- stage 3x18x128 f16 input patch into LDS (54 pixels * 16 chunks) ----
  for (int chunk = tid; chunk < 54 * 16; chunk += 256) {
    int pix  = chunk >> 4;                   // 0..53
    int part = chunk & 15;                   // 16B piece of a pixel's 256B
    int row = pix / 18, col = pix - row * 18;
    int ix = oxb + col;
    if (ix > Win - 1) ix = Win - 1;          // only feeds masked lanes
    const h8* g = (const h8*)(act + (((size_t)n * Hin + oy + row) * Win + ix) * CMID
                                  + part * 8);
    *(h8*)(&patch[(row * 18 + col) * PADW + part * 8]) = *g;
  }
  __syncthreads();

  int ox = oxb + lm;
  const bool valid = ox < Wo;
  if (!valid) ox = Wo - 1;

  const _Float16* wrow = wt + (size_t)(cobase + lm) * KTOT + hf * 8;

  f8 acc0 = {}; f8 acc1 = {};
  for (int kk = 0; kk < KTOT; kk += 64) {
#pragma unroll
    for (int u = 0; u < 2; ++u) {
      const int k0   = kk + u * 32;
      const int kpos = k0 >> 7;
      const int ky = kpos / 3, kx = kpos - ky * 3;
      const int cib = k0 & 127;
      // A: weights from global (b128 x2)
      h8 a0 = *(const h8*)(wrow + k0);
      h8 a1 = *(const h8*)(wrow + k0 + 16);
      h16 A = __builtin_shufflevector(a0, a1, 0,1,2,3,4,5,6,7,8,9,10,11,12,13,14,15);
      // B: activations from LDS patch (ds b128 x2)
      const _Float16* bp = &patch[(ky * 18 + (lm + kx)) * PADW + cib + hf * 8];
      h8 b0 = *(const h8*)bp;
      h8 b1 = *(const h8*)(bp + 16);
      h16 B = __builtin_shufflevector(b0, b1, 0,1,2,3,4,5,6,7,8,9,10,11,12,13,14,15);
      if (u == 0)
        acc0 = __builtin_amdgcn_wmma_f32_16x16x32_f16(false, A, false, B, (short)0, acc0, false, false);
      else
        acc1 = __builtin_amdgcn_wmma_f32_16x16x32_f16(false, A, false, B, (short)0, acc1, false, false);
    }
  }
  f8 acc = acc0 + acc1;
  if (valid) {
    const int cb = cobase + hf * 8;          // D: vgpr r -> M = r + 8*half
    float* op = out + (((size_t)n * Ho + oy) * Wo + ox) * CMID + cb;
#pragma unroll
    for (int rr = 0; rr < 8; ++rr) op[rr] = acc[rr] + bias[cb + rr];
  }
}

// ---------------------------------------------------------------------------
// 3x3 stride-2 maxpool, NHWC f32
// ---------------------------------------------------------------------------
__global__ __launch_bounds__(256) void k_maxpool(
    const float* __restrict__ in, float* __restrict__ out,
    int Hc, int Wc, int Hp, int Wp, int total)
{
  int idx = blockIdx.x * 256 + threadIdx.x;
  if (idx >= total) return;
  int c = idx & 127;
  int p = idx >> 7;
  int x = p % Wp; p /= Wp;
  int y = p % Hp;
  int n = p / Hp;
  const float* bp = in + (((size_t)n * Hc + 2 * y) * Wc + 2 * x) * CMID + c;
  float m = -3.402823466e38f;
#pragma unroll
  for (int dy = 0; dy < 3; ++dy)
#pragma unroll
    for (int dx = 0; dx < 3; ++dx) {
      float v = bp[((size_t)dy * Wc + dx) * CMID];
      m = fmaxf(m, v);
    }
  out[idx] = m;
}

// ---------------------------------------------------------------------------
// BN stats: per-channel sum / sumsq (NHWC, c = idx & 127)
// ---------------------------------------------------------------------------
__global__ __launch_bounds__(256) void k_zero_stats(float* s) { s[threadIdx.x] = 0.f; }

__global__ __launch_bounds__(256) void k_bn_stats(
    const float* __restrict__ in, float* __restrict__ stats, size_t total)
{
  __shared__ float ssum[256], ssq[256];
  const int t = threadIdx.x;
  float s = 0.f, q = 0.f;
  const size_t stride = (size_t)gridDim.x * 256;
  for (size_t i = (size_t)blockIdx.x * 256 + t; i < total; i += stride) {
    float v = in[i];
    s += v; q += v * v;
  }
  ssum[t] = s; ssq[t] = q;
  __syncthreads();
  if (t < 128) {
    atomicAdd(&stats[t],       ssum[t] + ssum[t + 128]);
    atomicAdd(&stats[128 + t], ssq[t]  + ssq[t + 128]);
  }
}

// ---------------------------------------------------------------------------
// BN (training-mode, biased var) + ReLU apply. Writes f16 act (next stage)
// or f32 (final stage).
// ---------------------------------------------------------------------------
__global__ __launch_bounds__(256) void k_bn_apply(
    const float* __restrict__ in, const float* __restrict__ stats,
    const float* __restrict__ g, const float* __restrict__ b,
    float cnt_inv, _Float16* __restrict__ out16, float* __restrict__ out32,
    int use_half, int total)
{
  int idx = blockIdx.x * 256 + threadIdx.x;
  if (idx >= total) return;
  int c = idx & 127;
  float mu  = stats[c] * cnt_inv;
  float var = stats[128 + c] * cnt_inv - mu * mu;
  float y = (in[idx] - mu) * rsqrtf(var + 1e-5f) * g[c] + b[c];
  y = fmaxf(y, 0.f);
  if (use_half) out16[idx] = (_Float16)y;
  else          out32[idx] = y;
}

// ---------------------------------------------------------------------------
// LRN(size=128, k=1, alpha=1e-4, beta=0.75) + FC(2048 -> 7), one block per n.
// act5: NHWC f32 [16][4][4][128]. Feature flatten order matches NCHW reshape:
// f = c*16 + (h*4+w).
// ---------------------------------------------------------------------------
__global__ __launch_bounds__(128) void k_lrn_fc(
    const float* __restrict__ act5, const float* __restrict__ Wr,
    const float* __restrict__ br, float* __restrict__ reg)
{
  __shared__ float sq[128], red[128], lrnv[2048];
  const int n = blockIdx.x, t = threadIdx.x;      // t = channel
  const float* ap = act5 + (size_t)n * 2048;
  float q = 0.f;
  for (int hw = 0; hw < 16; ++hw) { float v = ap[hw * 128 + t]; q += v * v; }
  sq[t] = q;
  __syncthreads();
  int lo = t - 64; if (lo < 0) lo = 0;
  int hi = t + 63; if (hi > 127) hi = 127;
  float win = 0.f;
  for (int d = lo; d <= hi; ++d) win += sq[d];
  float denom = powf(1.0f + 1e-4f * win / 128.0f, 0.75f);
  for (int hw = 0; hw < 16; ++hw)
    lrnv[t * 16 + hw] = ap[hw * 128 + t] / denom;  // f = c*16 + hw
  __syncthreads();
  for (int j = 0; j < 7; ++j) {
    float p = 0.f;
    for (int f = t; f < 2048; f += 128) p += lrnv[f] * Wr[j * 2048 + f];
    red[t] = p;
    __syncthreads();
    for (int off = 64; off; off >>= 1) {
      if (t < off) red[t] += red[t + off];
      __syncthreads();
    }
    if (t == 0) reg[n * 7 + j] = red[0] + br[j];
    __syncthreads();
  }
}

// ---------------------------------------------------------------------------
// Spectral norm: sequential scan over the batch (u,v carried), 4 power
// iterations per 2x3 matrix; also m = sigmoid(reg[:,6]).
// ---------------------------------------------------------------------------
__global__ void k_specnorm(const float* __restrict__ reg,
                           float* __restrict__ theta, float* __restrict__ m,
                           const float* __restrict__ u0, const float* __restrict__ v0)
{
  float u[2] = {u0[0], u0[1]};
  float v[3] = {v0[0], v0[1], v0[2]};
  for (int n = 0; n < 16; ++n) {
    const float* W = reg + n * 7;          // W[i*3+j]
    for (int it = 0; it < 4; ++it) {
      float nv[3], nn = 0.f;
      for (int j = 0; j < 3; ++j) { nv[j] = W[j] * u[0] + W[3 + j] * u[1]; nn += nv[j] * nv[j]; }
      nn = fmaxf(sqrtf(nn), 1e-12f);
      for (int j = 0; j < 3; ++j) v[j] = nv[j] / nn;
      float nu[2]; nn = 0.f;
      for (int i = 0; i < 2; ++i) {
        nu[i] = W[i * 3] * v[0] + W[i * 3 + 1] * v[1] + W[i * 3 + 2] * v[2];
        nn += nu[i] * nu[i];
      }
      nn = fmaxf(sqrtf(nn), 1e-12f);
      for (int i = 0; i < 2; ++i) u[i] = nu[i] / nn;
    }
    float Wv0 = W[0] * v[0] + W[1] * v[1] + W[2] * v[2];
    float Wv1 = W[3] * v[0] + W[4] * v[1] + W[5] * v[2];
    float sigma = u[0] * Wv0 + u[1] * Wv1;
    for (int k = 0; k < 6; ++k) theta[n * 6 + k] = W[k] / sigma;
    m[n] = 1.f / (1.f + expf(-W[6]));
  }
}

// ---------------------------------------------------------------------------
// affine_grid (align_corners=False) + reflect grid_sample + scale by m.
// x: NCHW f32 in, out: NCHW f32.
// ---------------------------------------------------------------------------
__device__ __forceinline__ float reflectf(float x, float size)
{
  x = fabsf(x + 0.5f);
  x = fmodf(x, 2.0f * size);
  if (x > size) x = 2.0f * size - x;
  return fminf(fmaxf(x - 0.5f, 0.0f), size - 1.0f);
}

__global__ __launch_bounds__(256) void k_sample(
    const float* __restrict__ x, const float* __restrict__ theta,
    const float* __restrict__ m, float* __restrict__ out, int total)
{
  int idx = blockIdx.x * 256 + threadIdx.x;
  if (idx >= total) return;                  // total = 16*224*224
  const int w = idx % 224;
  const int h = (idx / 224) % 224;
  const int n = idx / (224 * 224);
  const float* th = theta + n * 6;
  const float gx = (2.f * w + 1.f) / 224.f - 1.f;
  const float gy = (2.f * h + 1.f) / 224.f - 1.f;
  const float sx = th[0] * gx + th[1] * gy + th[2];
  const float sy = th[3] * gx + th[4] * gy + th[5];
  const float ix = reflectf(((sx + 1.f) * 224.f - 1.f) * 0.5f, 224.f);
  const float iy = reflectf(((sy + 1.f) * 224.f - 1.f) * 0.5f, 224.f);
  const float x0 = floorf(ix), y0 = floorf(iy);
  const float wx = ix - x0, wy = iy - y0;
  int x0i = (int)x0, y0i = (int)y0;
  int x1i = x0i + 1,  y1i = y0i + 1;
  x0i = min(max(x0i, 0), 223); x1i = min(max(x1i, 0), 223);
  y0i = min(max(y0i, 0), 223); y1i = min(max(y1i, 0), 223);
  const float w00 = (1.f - wx) * (1.f - wy), w01 = wx * (1.f - wy);
  const float w10 = (1.f - wx) * wy,         w11 = wx * wy;
  const float mm = m[n];
  for (int c = 0; c < 16; ++c) {
    const float* im = x + ((size_t)(n * 16 + c)) * 224 * 224;
    float v = im[y0i * 224 + x0i] * w00 + im[y0i * 224 + x1i] * w01
            + im[y1i * 224 + x0i] * w10 + im[y1i * 224 + x1i] * w11;
    out[((size_t)(n * 16 + c) * 224 + h) * 224 + w] = v * mm;
  }
}

// ---------------------------------------------------------------------------
extern "C" void kernel_launch(void* const* d_in, const int* in_sizes, int n_in,
                              void* d_out, int out_size, void* d_ws, size_t ws_size,
                              hipStream_t stream)
{
  (void)in_sizes; (void)n_in; (void)out_size; (void)ws_size;
  const float* x       = (const float*)d_in[0];
  const float* conv0_w = (const float*)d_in[1];
  const float* conv0_b = (const float*)d_in[2];
  const float* convs_w = (const float*)d_in[3];
  const float* convs_b = (const float*)d_in[4];
  const float* bn_g    = (const float*)d_in[5];
  const float* bn_b    = (const float*)d_in[6];
  const float* Wr      = (const float*)d_in[7];
  const float* br      = (const float*)d_in[8];
  const float* u0      = (const float*)d_in[9];
  const float* v0      = (const float*)d_in[10];

  char* ws = (char*)d_ws;
  size_t off = 0;
  auto take = [&](size_t bytes) -> size_t {
    size_t o = off;
    off += (bytes + 255) & ~(size_t)255;
    return o;
  };
  _Float16* ACT  = (_Float16*)(ws + take((size_t)16 * 224 * 224 * 128 * 2)); // f16 NHWC acts
  float*    CONV = (float*)   (ws + take((size_t)16 * 224 * 224 * 128 * 4)); // f32 conv out
  float*    POOL = (float*)   (ws + take((size_t)16 * 110 * 110 * 128 * 4)); // f32 pool out
  _Float16* W16  = (_Float16*)(ws + take((size_t)5 * 128 * KTOT * 2));       // f16 weights
  float*    A5   = (float*)   (ws + take((size_t)16 * 4 * 4 * 128 * 4));     // stage-5 act f32
  float*    STAT = (float*)   (ws + take(256 * 4));
  float*    REG  = (float*)   (ws + take(16 * 7 * 4));
  float*    TH   = (float*)   (ws + take(16 * 6 * 4));
  float*    M    = (float*)   (ws + take(16 * 4));

  // weights -> f16 GEMM layout
  {
    int total = 5 * 128 * KTOT;
    k_prep_w<<<(total + 255) / 256, 256, 0, stream>>>(convs_w, W16, total);
  }

  // conv0 + BN + ReLU -> ACT (f16)
  {
    int total = 16 * 224 * 224 * 128;
    k_conv0<<<(total + 255) / 256, 256, 0, stream>>>(x, conv0_w, conv0_b, CONV, total);
    k_zero_stats<<<1, 256, 0, stream>>>(STAT);
    k_bn_stats<<<4096, 256, 0, stream>>>(CONV, STAT, (size_t)total);
    float cnt_inv = 1.0f / (16.0f * 224.0f * 224.0f);
    k_bn_apply<<<(total + 255) / 256, 256, 0, stream>>>(
        CONV, STAT, bn_g, bn_b, cnt_inv, ACT, nullptr, 1, total);
  }

  const int HinT[5] = {224, 110, 53, 25, 11};
  for (int s = 0; s < 5; ++s) {
    const int Hin = HinT[s], Win = Hin;
    const int Ho = Hin - 2, Wo = Ho;
    const int Hp = (Ho - 3) / 2 + 1, Wp = Hp;
    const int tpr = (Wo + 15) / 16;

    k_conv3x3_wmma<<<16 * Ho * tpr, 256, 0, stream>>>(
        ACT, W16 + (size_t)s * 128 * KTOT, convs_b + s * 128, CONV, Hin, Win, Ho, Wo);

    int ptotal = 16 * Hp * Wp * 128;
    k_maxpool<<<(ptotal + 255) / 256, 256, 0, stream>>>(CONV, POOL, Ho, Wo, Hp, Wp, ptotal);

    k_zero_stats<<<1, 256, 0, stream>>>(STAT);
    int sb = (ptotal + 255) / 256; if (sb > 4096) sb = 4096;
    k_bn_stats<<<sb, 256, 0, stream>>>(POOL, STAT, (size_t)ptotal);

    float cnt_inv = 1.0f / (16.0f * (float)Hp * (float)Wp);
    if (s < 4)
      k_bn_apply<<<(ptotal + 255) / 256, 256, 0, stream>>>(
          POOL, STAT, bn_g + (s + 1) * 128, bn_b + (s + 1) * 128, cnt_inv, ACT, nullptr, 1, ptotal);
    else
      k_bn_apply<<<(ptotal + 255) / 256, 256, 0, stream>>>(
          POOL, STAT, bn_g + (s + 1) * 128, bn_b + (s + 1) * 128, cnt_inv, nullptr, A5, 0, ptotal);
  }

  // LRN + FC
  k_lrn_fc<<<16, 128, 0, stream>>>(A5, Wr, br, REG);

  // spectral norm scan + sigmoid gate
  k_specnorm<<<1, 1, 0, stream>>>(REG, TH, M, u0, v0);

  // affine grid + reflect bilinear sample + gate
  {
    int total = 16 * 224 * 224;
    k_sample<<<(total + 255) / 256, 256, 0, stream>>>(x, TH, M, (float*)d_out, total);
  }
}
